// sLSTMCell_36507222016550
// MI455X (gfx1250) — matrix operand order repaired and checked by reading, not compiled, not run on hardware
//
#include <hip/hip_runtime.h>
#include <hip/hip_bf16.h>

// sLSTM fused kernel for gfx1250 (MI455X), double-buffered LDS pipeline.
// gates = [x|h] @ [Wx|Wh]^T + b on v_wmma_f32_16x16x32_bf16 (f32 accumulate),
// exp-gate sLSTM update fused into the epilogue (gates never touch HBM).

#define BATCH 16384
#define INDIM 1024
#define HDIM  1024

typedef __attribute__((ext_vector_type(16))) __bf16 v16bf;
typedef __attribute__((ext_vector_type(8)))  float  v8f;

union Frag16 { v16bf v; uint4 q[2]; };
union Ld16   { float4 f4[4]; float f[16]; };
union Pk16   { __bf16 h[16]; uint4 q[2]; };

// LDS row stride in ushort units: 32 bf16 + 8 pad = 80 bytes (bank-spread for b128)
#define LROW 40

__device__ __forceinline__ void load_tile(const float* ap, const float* bp,
                                          int kk, int ldSub, Ld16& a16, Ld16& b16) {
    const float4* a4 = (const float4*)(ap + kk + ldSub);
    const float4* b4 = (const float4*)(bp + kk + ldSub);
    #pragma unroll
    for (int e = 0; e < 4; ++e) { a16.f4[e] = a4[e]; b16.f4[e] = b4[e]; }
}

__device__ __forceinline__ void stage_tile(const Ld16& a16, const Ld16& b16,
                                           unsigned short* __restrict__ A,
                                           unsigned short* __restrict__ Bl,
                                           int ldRow, int ldSub) {
    Pk16 pa, pb;
    #pragma unroll
    for (int e = 0; e < 16; ++e) {
        pa.h[e] = (__bf16)a16.f[e];
        pb.h[e] = (__bf16)b16.f[e];
    }
    uint4* as = (uint4*)&A[ldRow * LROW + ldSub];
    uint4* bs = (uint4*)&Bl[ldRow * LROW + ldSub];
    as[0] = pa.q[0]; as[1] = pa.q[1];
    bs[0] = pb.q[0]; bs[1] = pb.q[1];
}

__device__ __forceinline__ void mma_step(const unsigned short* __restrict__ A,
                                         const unsigned short* __restrict__ Bl,
                                         int wr, int wc, int lane, v8f acc[4][2]) {
    // A fragments: 16x32 bf16; lanes 0-15 hold k{0..7,16..23}, lanes 16-31 k{8..15,24..31}
    Frag16 af[2];
    #pragma unroll
    for (int s = 0; s < 2; ++s) {
        const int mrow = wr * 32 + s * 16 + (lane & 15);
        const int kb   = (lane < 16) ? 0 : 8;
        af[s].q[0] = *(const uint4*)&A[mrow * LROW + kb];
        af[s].q[1] = *(const uint4*)&A[mrow * LROW + kb + 16];
    }
    // B fragments: 32x16 bf16; lanes 0-15 hold k{0..7,16..23}, lanes 16-31 k{8..15,24..31}
    Frag16 bf[4];
    #pragma unroll
    for (int g = 0; g < 4; ++g) {
        const int trow = g * 32 + wc * 16 + (lane & 15);
        const int kb   = (lane < 16) ? 0 : 8;
        const uint4* p = (const uint4*)&Bl[trow * LROW + kb];
        bf[g].q[0] = p[0]; bf[g].q[1] = p[2];
    }
    #pragma unroll
    for (int g = 0; g < 4; ++g)
        #pragma unroll
        for (int s = 0; s < 2; ++s)
            acc[g][s] = __builtin_amdgcn_wmma_f32_16x16x32_bf16(
                false, af[s].v, false, bf[g].v,
                (short)0, acc[g][s], false, false);
}

__global__ __launch_bounds__(256) void slstm_fused(
    const float* __restrict__ X,  const float* __restrict__ Hst,
    const float* __restrict__ C,  const float* __restrict__ N,
    const float* __restrict__ M,  const float* __restrict__ Wx,
    const float* __restrict__ Wh, const float* __restrict__ Bb,
    float* __restrict__ Out)
{
    __shared__ unsigned short Alds[2][128 * LROW];   // 2 x (128 rows x 32 k bf16)
    __shared__ unsigned short Blds[2][128 * LROW];   // 2 x (4 gates x 32 cols x 32 k)

    const int tid  = threadIdx.x;
    const int lane = tid & 31;
    const int wave = tid >> 5;
    const int wr   = wave >> 1;        // 0..3 : 32-row group
    const int wc   = wave & 1;         // 0..1 : 16-col group in the 32-col tile
    const int rowBase = blockIdx.y * 128;
    const int jBase   = blockIdx.x * 32;

    // global load assignment: 2 threads per tile row, 16 consecutive f32 each
    const int ldRow = tid >> 1;            // 0..127
    const int ldSub = (tid & 1) * 16;      // k sub-chunk 0 or 16
    const int bg    = ldRow >> 5;          // gate index of B-tile row
    const int bj    = ldRow & 31;          // col-in-gate of B-tile row
    const float* Arow_x = X   + (size_t)(rowBase + ldRow) * INDIM;
    const float* Arow_h = Hst + (size_t)(rowBase + ldRow) * HDIM;
    const float* Brow_x = Wx  + (size_t)(bg * HDIM + jBase + bj) * INDIM;
    const float* Brow_h = Wh  + (size_t)(bg * HDIM + jBase + bj) * HDIM;

    v8f acc[4][2] = {};                    // [gate][row subtile]

    // -------- pipeline prologue: stage tile 0 --------
    Ld16 a16, b16;
    load_tile(Arow_x, Brow_x, 0, ldSub, a16, b16);
    stage_tile(a16, b16, Alds[0], Blds[0], ldRow, ldSub);
    __syncthreads();

    // -------- steady state: one barrier per K-step, next-tile loads overlap WMMA --------
    for (int kt = 0; kt < 63; ++kt) {
        const int cur = kt & 1, nxt = cur ^ 1;
        const int k1 = (kt + 1) * 32;
        const float* ap; const float* bp; int kk;
        if (k1 < 1024) { ap = Arow_x; bp = Brow_x; kk = k1;        }
        else           { ap = Arow_h; bp = Brow_h; kk = k1 - 1024; }
        load_tile(ap, bp, kk, ldSub, a16, b16);          // global loads in flight
        mma_step(Alds[cur], Blds[cur], wr, wc, lane, acc);
        stage_tile(a16, b16, Alds[nxt], Blds[nxt], ldRow, ldSub);
        __syncthreads();
    }
    mma_step(Alds[1], Blds[1], wr, wc, lane, acc);       // kt = 63

    // ---------------- fused sLSTM epilogue (branch-free) ----------------
    const int j = jBase + wc * 16 + (lane & 15);
    const float b_i = Bb[0 * HDIM + j];
    const float b_f = Bb[1 * HDIM + j];
    const float b_z = Bb[2 * HDIM + j];
    const float b_o = Bb[3 * HDIM + j];
    const size_t BH = (size_t)BATCH * HDIM;

    #pragma unroll
    for (int s = 0; s < 2; ++s) {
        const int rbase = rowBase + wr * 32 + s * 16 + ((lane < 16) ? 0 : 8);
        #pragma unroll
        for (int r = 0; r < 8; ++r) {
            const size_t idx = (size_t)(rbase + r) * HDIM + j;
            const float it = acc[0][s][r] + b_i;
            const float ft = acc[1][s][r] + b_f;
            const float zt = acc[2][s][r] + b_z;
            const float ot = acc[3][s][r] + b_o;
            const float mo = __builtin_nontemporal_load(M + idx);
            const float co = __builtin_nontemporal_load(C + idx);
            const float no = __builtin_nontemporal_load(N + idx);
            const float mn = fmaxf(ft + mo, it);
            const float fg = __expf(ft + mo - mn);
            const float ig = __expf(it - mn);
            // tanh(z) = 1 - 2/(e^{2z}+1): saturates correctly at +/-inf, no branches
            const float zz = 1.0f - __fdividef(2.0f, __expf(2.0f * zt) + 1.0f);
            const float oo = __fdividef(1.0f, 1.0f + __expf(-ot));
            const float cn = fg * co + ig * zz;
            const float nn = fg * no + ig;
            const float hn = oo * __fdividef(cn, fmaxf(fabsf(nn), 1.0f));
            __builtin_nontemporal_store(hn, Out + idx);
            __builtin_nontemporal_store(cn, Out + BH + idx);
            __builtin_nontemporal_store(nn, Out + 2 * BH + idx);
            __builtin_nontemporal_store(mn, Out + 3 * BH + idx);
        }
    }
}

extern "C" void kernel_launch(void* const* d_in, const int* in_sizes, int n_in,
                              void* d_out, int out_size, void* d_ws, size_t ws_size,
                              hipStream_t stream) {
    (void)in_sizes; (void)n_in; (void)out_size; (void)d_ws; (void)ws_size;
    const float* x  = (const float*)d_in[0];
    const float* h  = (const float*)d_in[1];
    const float* c  = (const float*)d_in[2];
    const float* n  = (const float*)d_in[3];
    const float* m  = (const float*)d_in[4];
    const float* Wx = (const float*)d_in[5];
    const float* Wh = (const float*)d_in[6];
    const float* bb = (const float*)d_in[7];
    float* out = (float*)d_out;

    dim3 grid(HDIM / 32, BATCH / 128);   // (32, 128) = 4096 workgroups
    slstm_fused<<<grid, 256, 0, stream>>>(x, h, c, n, m, Wx, Wh, bb, out);
}
